// FullGAT_21981642621099
// MI455X (gfx1250) — matrix operand order repaired and checked by reading, not compiled, not run on hardware
//
#include <hip/hip_runtime.h>

#define NP 100000
#define NF 50000
#define EL 800000
#define EC 200000
#define ER 200000
#define H  128
#define DIN 64

typedef _Float16 v16h __attribute__((ext_vector_type(16)));
typedef float    v8f  __attribute__((ext_vector_type(8)));

__device__ __forceinline__ int enc_f32(float f){
  int i = __float_as_int(f); return i >= 0 ? i : (i ^ 0x7fffffff);
}
__device__ __forceinline__ float dec_f32(int i){
  return __int_as_float(i >= 0 ? i : (i ^ 0x7fffffff));
}

// ---------------- fills ----------------
__global__ void fill_f32_k(float* __restrict__ p, float v, int n){
  for (int i = blockIdx.x*blockDim.x + threadIdx.x; i < n; i += gridDim.x*blockDim.x) p[i] = v;
}
__global__ void fill_i32_k(int* __restrict__ p, int v, int n){
  for (int i = blockIdx.x*blockDim.x + threadIdx.x; i < n; i += gridDim.x*blockDim.x) p[i] = v;
}

// -------- pack f32 weight [K x 128] into f16 WMMA B-fragment order --------
// fragment (kk, nt): lane L holds col N = nt*16 + (L&15), elements e=0..15 at
// K = kk*32 + (L>>4)*16 + e, stored contiguously (16 halves per lane).
__global__ void pack_w_k(const float* __restrict__ W, _Float16* __restrict__ dst, int K){
  int t = blockIdx.x*256 + threadIdx.x;
  if (t >= K*H) return;
  int e  = t & 15;
  int L  = (t >> 4) & 31;
  int nt = (t >> 9) & 7;
  int kk = t >> 12;
  int k  = kk*32 + (L >> 4)*16 + e;
  int n  = nt*16 + (L & 15);
  dst[t] = (_Float16)W[k*H + n];
}

// -------- WMMA GEMM: Out[N x 128] = X[N x K] @ W + bias (opt ReLU) --------
template<bool RELU, int K>
__global__ void gemm_k(const float* __restrict__ X, const _Float16* __restrict__ Wp,
                       const float* __restrict__ bias, float* __restrict__ Out, int N){
  const int wave = threadIdx.x >> 5;
  const int lane = threadIdx.x & 31;
  const int r0 = blockIdx.x*128 + wave*16;
  if (r0 >= N) return;                    // wave-uniform; N % 16 == 0 so no partial tiles
  const int g = lane >> 4;                // K-group (A) / row-group (C)
  const int m = lane & 15;                // row within A tile, col within B/C tile
  const int row = r0 + m;

  v8f acc[8] = {};                        // 8 C tiles of 16x16 (full 128 cols)

  #pragma unroll
  for (int kk = 0; kk < (K >> 5); ++kk){
    // A fragment contiguous: K = kk*32 + g*16 + e
    const float* xr = X + (size_t)row*K + kk*32 + g*16;
    float4 f0 = *(const float4*)(xr);
    float4 f1 = *(const float4*)(xr + 4);
    float4 f2 = *(const float4*)(xr + 8);
    float4 f3 = *(const float4*)(xr + 12);
    v16h a;
    a[0]=(_Float16)f0.x;  a[1]=(_Float16)f0.y;  a[2]=(_Float16)f0.z;  a[3]=(_Float16)f0.w;
    a[4]=(_Float16)f1.x;  a[5]=(_Float16)f1.y;  a[6]=(_Float16)f1.z;  a[7]=(_Float16)f1.w;
    a[8]=(_Float16)f2.x;  a[9]=(_Float16)f2.y;  a[10]=(_Float16)f2.z; a[11]=(_Float16)f2.w;
    a[12]=(_Float16)f3.x; a[13]=(_Float16)f3.y; a[14]=(_Float16)f3.z; a[15]=(_Float16)f3.w;

    const _Float16* wb = Wp + (size_t)kk*8*512;   // (kk*8+nt)*32 lanes *16 halves
    #pragma unroll
    for (int nt = 0; nt < 8; ++nt){
      v16h b = *(const v16h*)(wb + (size_t)nt*512 + lane*16);
      acc[nt] = __builtin_amdgcn_wmma_f32_16x16x32_f16(
          false, a, false, b, (short)0, acc[nt], false, false);
    }
  }

  #pragma unroll
  for (int nt = 0; nt < 8; ++nt){
    int col = nt*16 + m;
    float bv = bias[col];
    #pragma unroll
    for (int v = 0; v < 8; ++v){
      int r = r0 + g*8 + v;               // C layout: lanes 0-15 -> M=v, 16-31 -> M=8+v
      float val = acc[nt][v] + bv;
      if (RELU) val = val > 0.0f ? val : 0.0f;
      Out[(size_t)r*H + col] = val;
    }
  }
}

// -------- edge phase --------
__global__ void edge_score_k(const float* __restrict__ hl, const float* __restrict__ hr,
                             const int* __restrict__ src, const int* __restrict__ dst,
                             const float* __restrict__ att, float* __restrict__ score,
                             int* __restrict__ mx, int E){
  int e = (blockIdx.x*blockDim.x + threadIdx.x) >> 5;
  int lane = threadIdx.x & 31;
  if (e >= E) return;
  int s = src[e], d = dst[e];
  float4 a4 = *(const float4*)(att + lane*4);
  float4 l4 = *(const float4*)(hl + (size_t)s*H + lane*4);
  float4 r4 = *(const float4*)(hr + (size_t)d*H + lane*4);
  float acc = 0.f, v;
  v = l4.x + r4.x; v = v > 0.f ? v : 0.2f*v; acc += v*a4.x;
  v = l4.y + r4.y; v = v > 0.f ? v : 0.2f*v; acc += v*a4.y;
  v = l4.z + r4.z; v = v > 0.f ? v : 0.2f*v; acc += v*a4.z;
  v = l4.w + r4.w; v = v > 0.f ? v : 0.2f*v; acc += v*a4.w;
  #pragma unroll
  for (int off = 16; off; off >>= 1) acc += __shfl_xor(acc, off, 32);
  if (lane == 0){ score[e] = acc; atomicMax(&mx[d], enc_f32(acc)); }
}

__global__ void edge_exp_k(float* __restrict__ score, const int* __restrict__ dst,
                           const int* __restrict__ mx, float* __restrict__ ssum, int E){
  int e = blockIdx.x*blockDim.x + threadIdx.x;
  if (e >= E) return;
  int d = dst[e];
  float a = expf(score[e] - dec_f32(mx[d]));
  score[e] = a;
  atomicAdd(&ssum[d], a);
}

__global__ void edge_agg_k(const float* __restrict__ score, const float* __restrict__ ssum,
                           const int* __restrict__ src, const int* __restrict__ dst,
                           const float* __restrict__ hl, float* __restrict__ agg, int E){
  int e = (blockIdx.x*blockDim.x + threadIdx.x) >> 5;
  int lane = threadIdx.x & 31;
  if (e >= E) return;
  int s = src[e], d = dst[e];
  float w = score[e] / (ssum[d] + 1e-16f);
  float4 h4 = *(const float4*)(hl + (size_t)s*H + lane*4);
  float* ap = agg + (size_t)d*H + lane*4;
  atomicAdd(ap + 0, w*h4.x);
  atomicAdd(ap + 1, w*h4.y);
  atomicAdd(ap + 2, w*h4.z);
  atomicAdd(ap + 3, w*h4.w);
}

__global__ void combine_pano_k(const float* __restrict__ agg, const float* __restrict__ b0,
                               const float* __restrict__ b1, float* __restrict__ out, int n){
  int i = blockIdx.x*blockDim.x + threadIdx.x;
  if (i >= n) return;
  int c = i & (H-1);
  float v = 0.5f*(agg[i] + b0[c] + b1[c]);
  out[i] = v > 0.f ? v : 0.f;
}

__global__ void combine_foot_k(const float* __restrict__ agg, const float* __restrict__ b2,
                               float* __restrict__ out, int n){
  int i = blockIdx.x*blockDim.x + threadIdx.x;
  if (i >= n) return;
  int c = i & (H-1);
  float v = agg[i] + b2[c];
  out[i] = v > 0.f ? v : 0.f;
}

// -------- M=1 heads --------
__global__ void lp_k(const float* __restrict__ xf, const float* __restrict__ lw,
                     const float* __restrict__ lb, const float* __restrict__ a,
                     const float* __restrict__ wz, const float* __restrict__ bz,
                     float* __restrict__ lp, int n){
  int row = (blockIdx.x*blockDim.x + threadIdx.x) >> 5;
  int lane = threadIdx.x & 31;
  if (row >= n) return;
  float s = xf[(size_t)row*DIN + lane*2]     * lw[lane*2]
          + xf[(size_t)row*DIN + lane*2 + 1] * lw[lane*2 + 1];
  float4 a4 = *(const float4*)(a + (size_t)row*H + lane*4);
  float4 w4 = *(const float4*)(wz + lane*4);
  s += a4.x*w4.x + a4.y*w4.y + a4.z*w4.z + a4.w*w4.w;
  #pragma unroll
  for (int off = 16; off; off >>= 1) s += __shfl_xor(s, off, 32);
  if (lane == 0) lp[row] = s + lb[0] + bz[0];
}

__global__ void final_k(const float* __restrict__ h, const float* __restrict__ w3,
                        const float* __restrict__ b3, const float* __restrict__ lp,
                        float* __restrict__ out, int n){
  int row = (blockIdx.x*blockDim.x + threadIdx.x) >> 5;
  int lane = threadIdx.x & 31;
  if (row >= n) return;
  float4 h4 = *(const float4*)(h + (size_t)row*H + lane*4);
  float4 w4 = *(const float4*)(w3 + lane*4);
  float s = h4.x*w4.x + h4.y*w4.y + h4.z*w4.z + h4.w*w4.w;
  #pragma unroll
  for (int off = 16; off; off >>= 1) s += __shfl_xor(s, off, 32);
  if (lane == 0) out[row] = lp[row] + s + b3[0];
}

extern "C" void kernel_launch(void* const* d_in, const int* in_sizes, int n_in,
                              void* d_out, int out_size, void* d_ws, size_t ws_size,
                              hipStream_t stream){
  (void)in_sizes; (void)n_in; (void)out_size; (void)ws_size;
  const float* x_pano = (const float*)d_in[0];
  const float* x_foot = (const float*)d_in[1];
  const int* ei_links = (const int*)d_in[2];
  const int* ei_cont  = (const int*)d_in[3];
  const int* ei_rev   = (const int*)d_in[4];
  const float* lWl[2]  = {(const float*)d_in[5],  (const float*)d_in[11]};
  const float* lbl[2]  = {(const float*)d_in[6],  (const float*)d_in[12]};
  const float* lWr[2]  = {(const float*)d_in[7],  (const float*)d_in[13]};
  const float* lbr[2]  = {(const float*)d_in[8],  (const float*)d_in[14]};
  const float* latt[2] = {(const float*)d_in[9],  (const float*)d_in[15]};
  const float* lbias[2]= {(const float*)d_in[10], (const float*)d_in[16]};
  const float* nm_lin_w=(const float*)d_in[17];
  const float* nm_lin_b=(const float*)d_in[18];
  const float* nm_w0=(const float*)d_in[19];
  const float* nm_b0=(const float*)d_in[20];
  const float* nm_wc=(const float*)d_in[21];
  const float* nm_bc=(const float*)d_in[22];
  const float* nm_wz=(const float*)d_in[23];
  const float* nm_bz=(const float*)d_in[24];
  const float* m_w1=(const float*)d_in[25];
  const float* m_b1=(const float*)d_in[26];
  const float* m_w2=(const float*)d_in[27];
  const float* m_b2=(const float*)d_in[28];
  const float* m_w3=(const float*)d_in[29];
  const float* m_b3=(const float*)d_in[30];

  char* wsB = (char*)d_ws; size_t off = 0;
  auto alloc = [&](size_t bytes)->void*{
    void* p = wsB + off; off += (bytes + 255) & ~(size_t)255; return p;
  };
  float* xpA  = (float*)alloc((size_t)NP*H*4);
  float* xpB  = (float*)alloc((size_t)NP*H*4);
  float* xfA  = (float*)alloc((size_t)NF*H*4);
  float* xfB  = (float*)alloc((size_t)NF*H*4);
  float* hl   = (float*)alloc((size_t)NP*H*4);
  float* hr   = (float*)alloc((size_t)NP*H*4);
  float* aggP = (float*)alloc((size_t)NP*H*4);
  float* aggF = (float*)alloc((size_t)NF*H*4);
  float* score= (float*)alloc((size_t)EL*4);
  int*   mx   = (int*)  alloc((size_t)NP*4);
  float* ssum = (float*)alloc((size_t)NP*4);
  float* lp   = (float*)alloc((size_t)NF*4);
  _Float16* packW = (_Float16*)alloc((size_t)16*128*H*2);

  auto slot = [&](int i){ return packW + (size_t)i*128*H; };
  auto pack = [&](const float* W, int s, int K){
    int total = K*H;
    pack_w_k<<<(total+255)/256, 256, 0, stream>>>(W, slot(s), K);
  };
  for (int r = 0; r < 3; ++r){
    pack(lWl[0] + (size_t)r*DIN*H, r,     DIN);
    pack(lWr[0] + (size_t)r*DIN*H, 3 + r, DIN);
    pack(lWl[1] + (size_t)r*H*H,   6 + r, H);
    pack(lWr[1] + (size_t)r*H*H,   9 + r, H);
  }
  pack(nm_w0, 12, DIN); pack(nm_wc, 13, H); pack(m_w1, 14, H); pack(m_w2, 15, H);

  auto gemm = [&](const float* X, const _Float16* Wp, const float* bias,
                  float* Out, int N, int K, bool relu){
    dim3 grid((N + 127)/128), block(256);
    if (K == DIN){
      if (relu) gemm_k<true, DIN><<<grid, block, 0, stream>>>(X, Wp, bias, Out, N);
      else      gemm_k<false,DIN><<<grid, block, 0, stream>>>(X, Wp, bias, Out, N);
    } else {
      if (relu) gemm_k<true, H><<<grid, block, 0, stream>>>(X, Wp, bias, Out, N);
      else      gemm_k<false,H><<<grid, block, 0, stream>>>(X, Wp, bias, Out, N);
    }
  };
  auto fillf = [&](float* p, float v, int n){
    int blocks = (n + 255)/256; if (blocks > 4096) blocks = 4096;
    fill_f32_k<<<blocks, 256, 0, stream>>>(p, v, n);
  };
  auto filli = [&](int* p, int v, int n){
    int blocks = (n + 255)/256; if (blocks > 4096) blocks = 4096;
    fill_i32_k<<<blocks, 256, 0, stream>>>(p, v, n);
  };

  // ---- NullModel on original x_foot (uses hl/hr as scratch) ----
  gemm(x_foot, slot(12), nm_b0, hl, NF, DIN, true);
  gemm(hl,     slot(13), nm_bc, hr, NF, H,   true);
  gemm(hr,     slot(13), nm_bc, hl, NF, H,   true);
  lp_k<<<(NF+7)/8, 256, 0, stream>>>(x_foot, nm_lin_w, nm_lin_b, hl, nm_wz, nm_bz, lp, NF);

  auto relation = [&](const float* xs, int Ns, const float* xd, int Nd, int K,
                      const _Float16* Wl, const float* bl,
                      const _Float16* Wr, const float* br,
                      const float* att, const int* ei, int E, float* agg){
    gemm(xs, Wl, bl, hl, Ns, K, false);
    gemm(xd, Wr, br, hr, Nd, K, false);
    filli(mx, (int)0x80000000, Nd);
    fillf(ssum, 0.f, Nd);
    edge_score_k<<<(E+7)/8,    256, 0, stream>>>(hl, hr, ei, ei + E, att, score, mx, E);
    edge_exp_k  <<<(E+255)/256,256, 0, stream>>>(score, ei + E, mx, ssum, E);
    edge_agg_k  <<<(E+7)/8,    256, 0, stream>>>(score, ssum, ei, ei + E, hl, agg, E);
  };

  // ---- 2 hetero GATv2 layers ----
  for (int li = 0; li < 2; ++li){
    int K = li ? H : DIN;
    const float* xp = li ? xpA : x_pano;
    const float* xf = li ? xfA : x_foot;
    float* xp_out = li ? xpB : xpA;
    float* xf_out = li ? xfB : xfA;
    int bWl = li ? 6 : 0, bWr = li ? 9 : 3;

    fillf(aggP, 0.f, NP*H);
    relation(xp, NP, xp, NP, K, slot(bWl+0), lbl[li],       slot(bWr+0), lbr[li],
             latt[li],       ei_links, EL, aggP);
    relation(xf, NF, xp, NP, K, slot(bWl+1), lbl[li] + H,   slot(bWr+1), lbr[li] + H,
             latt[li] + H,   ei_cont,  EC, aggP);
    combine_pano_k<<<(NP*H+255)/256, 256, 0, stream>>>(aggP, lbias[li], lbias[li] + H,
                                                       xp_out, NP*H);
    fillf(aggF, 0.f, NF*H);
    relation(xp, NP, xf, NF, K, slot(bWl+2), lbl[li] + 2*H, slot(bWr+2), lbr[li] + 2*H,
             latt[li] + 2*H, ei_rev,   ER, aggF);
    combine_foot_k<<<(NF*H+255)/256, 256, 0, stream>>>(aggF, lbias[li] + 2*H, xf_out, NF*H);
  }

  // ---- final MLP head on footprints ----
  gemm(xfB, slot(14), m_b1, hl, NF, H, true);
  gemm(hl,  slot(15), m_b2, hr, NF, H, true);
  final_k<<<(NF+7)/8, 256, 0, stream>>>(hr, m_w3, m_b3, lp, (float*)d_out, NF);
}
